// HPEncoder_84705345011915
// MI455X (gfx1250) — matrix operand order repaired and checked by reading, not compiled, not run on hardware
//
#include <hip/hip_runtime.h>
#include <hip/hip_bf16.h>

typedef __attribute__((ext_vector_type(16))) _Float16 v16h;
typedef __attribute__((ext_vector_type(8)))  float    v8f;
typedef __attribute__((ext_vector_type(4)))  float    f32x4;

// ---------------------------------------------------------------------------
// zero workspace region
// ---------------------------------------------------------------------------
__global__ void zero_kernel(float* __restrict__ p, size_t n) {
    size_t i = (size_t)blockIdx.x * blockDim.x + threadIdx.x;
    size_t stride = (size_t)gridDim.x * blockDim.x;
    for (; i < n; i += stride) p[i] = 0.0f;
}

// ---------------------------------------------------------------------------
// scatter points: grid[i,j,k,:] += feats ; occ[i,j,k] = 1
// ---------------------------------------------------------------------------
__global__ void scatter_kernel(const int* __restrict__ coords,
                               const float* __restrict__ feats,
                               float* __restrict__ grid,
                               float* __restrict__ occ, int N) {
    int t = blockIdx.x * blockDim.x + threadIdx.x;
    if (t >= N) return;
    int i = coords[3 * t + 0];
    int j = coords[3 * t + 1];
    int k = coords[3 * t + 2];
    size_t v = ((size_t)i * 96 + j) * 96 + k;
    const float* f = feats + (size_t)8 * t;
    float* g = grid + v * 8;
#pragma unroll
    for (int c = 0; c < 8; ++c) atomicAdd(g + c, f[c]);
    occ[v] = 1.0f;
}

// ---------------------------------------------------------------------------
// 2x2x2 max-pool of occupancy
// ---------------------------------------------------------------------------
template <int DOUT>
__global__ void pool_kernel(const float* __restrict__ in, float* __restrict__ out) {
    constexpr int DIN = DOUT * 2;
    int t = blockIdx.x * blockDim.x + threadIdx.x;
    if (t >= DOUT * DOUT * DOUT) return;
    int x = t % DOUT;
    int y = (t / DOUT) % DOUT;
    int z = t / (DOUT * DOUT);
    float mx = 0.0f;
#pragma unroll
    for (int dz = 0; dz < 2; ++dz)
#pragma unroll
        for (int dy = 0; dy < 2; ++dy)
#pragma unroll
            for (int dx = 0; dx < 2; ++dx) {
                size_t idx = ((size_t)(z * 2 + dz) * DIN + (y * 2 + dy)) * DIN + (x * 2 + dx);
                float v = in[idx];
                mx = v > mx ? v : mx;
            }
    out[t] = mx;
}

// ---------------------------------------------------------------------------
// Implicit-GEMM conv + bias + ReLU + occupancy mask using V_WMMA_F32_16X16X32_F16.
//
// Block = 256 threads = 8 waves. All waves share one N-tile (16 output chans);
// wave w handles M-tile (blockIdx.x*8 + w) of 16 output voxels.
//
// Weights for the block's N-tile are staged ONCE into LDS, converted to f16,
// laid out in WMMA B-fragment order so each lane's per-chunk fetch is one
// contiguous 32-byte ds_load.
//
// Reduction axis q = tap*CIN + cin, chunked by 32 (packs taps when CIN < 32).
// ---------------------------------------------------------------------------
template <int CIN, int COUT, int KSZ, int STRIDE, int PAD, int DIN, int DOUT>
__global__ __launch_bounds__(256)
void conv_wmma_kernel(const float* __restrict__ x, const float* __restrict__ W,
                      const float* __restrict__ bias, const float* __restrict__ occ,
                      float* __restrict__ y) {
    constexpr int TAPS = KSZ * KSZ * KSZ;
    constexpr int KTOT = TAPS * CIN;          // full reduction length
    constexpr int NCH  = (KTOT + 31) / 32;    // K=32 chunks
    constexpr int MTOT = DOUT * DOUT * DOUT;

    // B fragments in LDS: element ((kc*2+half)*16 + n_local)*16 + i  (f16)
    __shared__ alignas(32) _Float16 wlds[NCH * 512];

    const int tid   = threadIdx.x;
    const int nBase = blockIdx.y * 16;

    // ---- cooperative weight staging: f32 -> f16, fragment-major ----
    for (int e = tid; e < NCH * 512; e += 256) {
        const int i  = e & 15;          // K element within fragment
        const int nl = (e >> 4) & 15;   // local output channel
        const int h  = (e >> 8) & 1;    // lane-half selector
        const int kc = e >> 9;          // K chunk
        const int q  = kc * 32 + (i >> 3) * 16 + h * 8 + (i & 7);
        float wv = 0.0f;
        if (q < KTOT) wv = W[(size_t)q * COUT + nBase + nl];
        wlds[e] = (_Float16)wv;
    }
    __syncthreads();

    const int wave = tid >> 5;
    const int lane = tid & 31;
    const int half = lane >> 4;       // K-subset selector (ISA 16-bit A/B layout)
    const int row  = lane & 15;       // A-matrix row / B,C column

    const int mBase = (blockIdx.x * 8 + wave) * 16;

    // This lane's A-row output voxel
    const int m  = mBase + row;
    const int xo = m % DOUT;
    const int yo = (m / DOUT) % DOUT;
    const int zo = m / (DOUT * DOUT);
    const int zi0 = zo * STRIDE - PAD;
    const int yi0 = yo * STRIDE - PAD;
    const int xi0 = xo * STRIDE - PAD;

    const int n = nBase + row;        // this lane's output channel (B/C column)

    const v16h* wfrag = reinterpret_cast<const v16h*>(wlds);

    v8f c = {};
    for (int kc = 0; kc < NCH; ++kc) {
        // ---- A fragment: 16 f16 per lane, two groups of 8 consecutive q ----
        v16h a;
#pragma unroll
        for (int g = 0; g < 2; ++g) {
            const int q0  = kc * 32 + g * 16 + half * 8;  // 8-aligned; single tap
            const int tap = q0 / CIN;
            const int ci  = q0 % CIN;
            const int dz = tap / (KSZ * KSZ);
            const int dy = (tap / KSZ) % KSZ;
            const int dx = tap % KSZ;
            const int zi = zi0 + dz, yi = yi0 + dy, xi = xi0 + dx;
            const bool ok = (tap < TAPS) &&
                            (zi >= 0) && (zi < DIN) &&
                            (yi >= 0) && (yi < DIN) &&
                            (xi >= 0) && (xi < DIN);
            f32x4 p0 = {}, p1 = {};
            if (ok) {
                const f32x4* s4 = reinterpret_cast<const f32x4*>(
                    x + ((size_t)((zi * DIN + yi) * DIN + xi)) * CIN + ci);
                p0 = s4[0];
                p1 = s4[1];
            }
#pragma unroll
            for (int e = 0; e < 4; ++e) {
                a[g * 8 + e]     = (_Float16)p0[e];
                a[g * 8 + 4 + e] = (_Float16)p1[e];
            }
        }
        // ---- B fragment: one contiguous 32B LDS read per lane ----
        v16h bf = wfrag[(kc * 2 + half) * 16 + row];

        // D = A*B + C  (f32 accumulate)
        c = __builtin_amdgcn_wmma_f32_16x16x32_f16(false, a, false, bf,
                                                   (short)0, c, false, false);
    }

    // ---- epilogue: bias + ReLU + occupancy mask ----
    const float bn = bias[n];
#pragma unroll
    for (int r = 0; r < 8; ++r) {
        const int mo = mBase + r + 8 * half;   // C/D layout: M = vgpr + 8*half
        if (mo < MTOT) {
            float v = c[r] + bn;
            v = v > 0.0f ? v : 0.0f;
            v *= occ[mo];
            y[(size_t)mo * COUT + n] = v;
        }
    }
}

// ---------------------------------------------------------------------------
// Host-side orchestration (graph-capture safe: launches only)
// ---------------------------------------------------------------------------
extern "C" void kernel_launch(void* const* d_in, const int* in_sizes, int n_in,
                              void* d_out, int out_size, void* d_ws, size_t ws_size,
                              hipStream_t stream) {
    const int*   coords = (const int*)  d_in[0];
    const float* feats  = (const float*)d_in[1];
    const float* W0  = (const float*)d_in[2];
    const float* b0  = (const float*)d_in[3];
    const float* Wd0 = (const float*)d_in[4];
    const float* bd0 = (const float*)d_in[5];
    const float* W1  = (const float*)d_in[6];
    const float* b1  = (const float*)d_in[7];
    const float* Wd1 = (const float*)d_in[8];
    const float* bd1 = (const float*)d_in[9];
    const float* W2  = (const float*)d_in[10];
    const float* b2  = (const float*)d_in[11];

    // workspace layout (floats)
    float* ws   = (float*)d_ws;
    float* g0   = ws;                               // 96^3 * 8
    float* occA = g0   + (size_t)884736 * 8;        // 96^3
    float* h0   = occA + 884736;                    // 96^3 * 16
    float* occ0 = h0   + (size_t)884736 * 16;       // 48^3
    float* h1   = occ0 + 110592;                    // 48^3 * 32
    float* occ1 = h1   + (size_t)110592 * 32;       // 24^3

    // outputs: (out2, out1, out0) concatenated
    float* out2 = (float*)d_out;                    // 24^3 * 64
    float* out1 = out2 + 884736;                    // 24^3 * 64
    float* out0 = out1 + 884736;                    // 48^3 * 32

    const int N = in_sizes[0] / 3;                  // 120000 points

    // 1) zero the scatter targets (grid + occupancy, contiguous)
    const size_t nz = (size_t)884736 * 8 + 884736;
    zero_kernel<<<2048, 256, 0, stream>>>(g0, nz);

    // 2) scatter points
    scatter_kernel<<<(N + 255) / 256, 256, 0, stream>>>(coords, feats, g0, occA, N);

    // 3) conv0: 3^3 s1, 8->16, 96^3, mask occA   (55296 M-tiles / 8 waves)
    conv_wmma_kernel<8, 16, 3, 1, 1, 96, 96>
        <<<dim3(6912, 1), 256, 0, stream>>>(g0, W0, b0, occA, h0);

    // 4) occ pool 96 -> 48
    pool_kernel<48><<<(110592 + 255) / 256, 256, 0, stream>>>(occA, occ0);

    // 5) convd0: 2^3 s2, 16->32, -> 48^3, mask occ0 -> out0
    conv_wmma_kernel<16, 32, 2, 2, 0, 96, 48>
        <<<dim3(864, 2), 256, 0, stream>>>(h0, Wd0, bd0, occ0, out0);

    // 6) conv1: 3^3 s1, 32->32, 48^3, mask occ0
    conv_wmma_kernel<32, 32, 3, 1, 1, 48, 48>
        <<<dim3(864, 2), 256, 0, stream>>>(out0, W1, b1, occ0, h1);

    // 7) occ pool 48 -> 24
    pool_kernel<24><<<(13824 + 255) / 256, 256, 0, stream>>>(occ0, occ1);

    // 8) convd1: 2^3 s2, 32->64, -> 24^3, mask occ1 -> out1
    conv_wmma_kernel<32, 64, 2, 2, 0, 48, 24>
        <<<dim3(108, 4), 256, 0, stream>>>(h1, Wd1, bd1, occ1, out1);

    // 9) conv2: 3^3 s1, 64->64, 24^3, mask occ1 -> out2
    conv_wmma_kernel<64, 64, 3, 1, 1, 24, 24>
        <<<dim3(108, 4), 256, 0, stream>>>(out1, W2, b2, occ1, out2);
}
